// RelativePositionSelfAttention_13477607375269
// MI455X (gfx1250) — matrix operand order repaired and checked by reading, not compiled, not run on hardware
//
#include <hip/hip_runtime.h>

// ---------------------------------------------------------------------------
// Relative-position self-attention, CDNA5 (gfx1250), wave32 + WMMA bf16.
// B=4, L=1024, D=1024, H=16, dk=64.
// ---------------------------------------------------------------------------

typedef __attribute__((ext_vector_type(16))) __bf16        v16bf;
typedef __attribute__((ext_vector_type(8)))  float         v8f;
typedef __attribute__((ext_vector_type(4)))  unsigned int  u32x4;
typedef unsigned short u16;

union FragB {
  u32x4 u[2];
  v16bf v;
  u16   s[16];
};

__device__ __forceinline__ u16 f2bf(float f) {
  unsigned int u = __float_as_uint(f);
  u += 0x7FFFu + ((u >> 16) & 1u);   // round-to-nearest-even
  return (u16)(u >> 16);
}

__device__ __forceinline__ v8f wmma_bf16(const FragB& a, const FragB& b, v8f c) {
  return __builtin_amdgcn_wmma_f32_16x16x32_bf16(false, a.v, false, b.v,
                                                 (short)0, c, false, false);
}

// A 16x32 bf16 fragment: lane half 0 -> K {koff..koff+7, koff+16..koff+23}
__device__ __forceinline__ void load_a(FragB& f, const u16* row, int k0, int hi) {
  const u16* p = row + k0 + hi * 8;
  f.u[0] = *(const u32x4*)(p);
  f.u[1] = *(const u32x4*)(p + 16);
}
// B 32x16 bf16 fragment: lane = column, K {hi*8..hi*8+7, 16+hi*8..16+hi*8+7}
__device__ __forceinline__ void load_b16(FragB& f, const u16* row, int k0, int hi) {
  const u16* p = row + k0 + hi * 8;
  f.u[0] = *(const u32x4*)(p);
  f.u[1] = *(const u32x4*)(p + 16);
}

constexpr int Bc  = 4;
constexpr int Lc  = 1024;
constexpr int Dc  = 1024;
constexpr int Hc  = 16;
constexpr int DKc = 64;

// ---------------------------------------------------------------------------
__global__ void cast_f32_to_bf16(const float* __restrict__ src, u16* __restrict__ dst,
                                 int n_src, int n_dst) {
  int i = blockIdx.x * blockDim.x + threadIdx.x;
  if (i < n_dst) dst[i] = (i < n_src) ? f2bf(src[i]) : (u16)0;
}

// ---------------------------------------------------------------------------
// out[m,n] = sum_k X[m,k]*W[n,k] + bias[n]      (i.e. X @ W^T + b)
// grid (M/32, N/64), block = 32 (one wave computes a 32x64 tile, 8 acc).
// mode 0: f32 [M,N];  mode 1: bf16 [B,H,L,dk];  mode 2: bf16 [B,H,dk,L] (V^T).
__global__ void __launch_bounds__(32)
gemm_bf16(const u16* __restrict__ X, const u16* __restrict__ W,
          const float* __restrict__ bias,
          float* __restrict__ outf, u16* __restrict__ outb,
          int M, int N, int Kd, int mode) {
  const int m0   = blockIdx.x * 32;
  const int n0   = blockIdx.y * 64;
  const int lane = threadIdx.x;
  const int r    = lane & 15;
  const int hi   = lane >> 4;

  const u16* xr0 = X + (size_t)(m0 + r) * Kd;
  const u16* xr1 = X + (size_t)(m0 + 16 + r) * Kd;
  const u16* wr0 = W + (size_t)(n0 + 0 * 16 + r) * Kd;
  const u16* wr1 = W + (size_t)(n0 + 1 * 16 + r) * Kd;
  const u16* wr2 = W + (size_t)(n0 + 2 * 16 + r) * Kd;
  const u16* wr3 = W + (size_t)(n0 + 3 * 16 + r) * Kd;

  v8f acc[8] = {};
  FragB ca0, ca1, cb0, cb1, cb2, cb3;   // current k-step fragments
  FragB na0, na1, nb0, nb1, nb2, nb3;   // next    k-step fragments

  auto load_cur = [&](int k0) {
    load_a(ca0, xr0, k0, hi);  load_a(ca1, xr1, k0, hi);
    load_b16(cb0, wr0, k0, hi); load_b16(cb1, wr1, k0, hi);
    load_b16(cb2, wr2, k0, hi); load_b16(cb3, wr3, k0, hi);
  };
  auto load_nxt = [&](int k0) {
    load_a(na0, xr0, k0, hi);  load_a(na1, xr1, k0, hi);
    load_b16(nb0, wr0, k0, hi); load_b16(nb1, wr1, k0, hi);
    load_b16(nb2, wr2, k0, hi); load_b16(nb3, wr3, k0, hi);
  };
  auto comp_cur = [&]() {
    acc[0] = wmma_bf16(ca0, cb0, acc[0]);
    acc[1] = wmma_bf16(ca0, cb1, acc[1]);
    acc[2] = wmma_bf16(ca0, cb2, acc[2]);
    acc[3] = wmma_bf16(ca0, cb3, acc[3]);
    acc[4] = wmma_bf16(ca1, cb0, acc[4]);
    acc[5] = wmma_bf16(ca1, cb1, acc[5]);
    acc[6] = wmma_bf16(ca1, cb2, acc[6]);
    acc[7] = wmma_bf16(ca1, cb3, acc[7]);
  };
  auto comp_nxt = [&]() {
    acc[0] = wmma_bf16(na0, nb0, acc[0]);
    acc[1] = wmma_bf16(na0, nb1, acc[1]);
    acc[2] = wmma_bf16(na0, nb2, acc[2]);
    acc[3] = wmma_bf16(na0, nb3, acc[3]);
    acc[4] = wmma_bf16(na1, nb0, acc[4]);
    acc[5] = wmma_bf16(na1, nb1, acc[5]);
    acc[6] = wmma_bf16(na1, nb2, acc[6]);
    acc[7] = wmma_bf16(na1, nb3, acc[7]);
  };

  // software-pipelined K loop (Kd divisible by 64)
  load_cur(0);
  for (int k0 = 0; k0 < Kd; k0 += 64) {
    load_nxt(k0 + 32);
    if (k0 + 256 < Kd) {   // stream W ahead of the pipeline (global_prefetch_b8)
      __builtin_prefetch(wr0 + k0 + 256, 0, 1);
      __builtin_prefetch(wr1 + k0 + 256, 0, 1);
      __builtin_prefetch(wr2 + k0 + 256, 0, 1);
      __builtin_prefetch(wr3 + k0 + 256, 0, 1);
    }
    comp_cur();
    if (k0 + 64 < Kd) load_cur(k0 + 64);
    comp_nxt();
  }

#pragma unroll
  for (int mt = 0; mt < 2; mt++) {
#pragma unroll
    for (int nt = 0; nt < 4; nt++) {
      const int n  = n0 + nt * 16 + r;
      const float bn = bias[n];
#pragma unroll
      for (int v = 0; v < 8; v++) {
        const int m   = m0 + mt * 16 + hi * 8 + v;   // C layout: VGPR v -> M=v+hi*8
        const float val = acc[mt * 4 + nt][v] + bn;
        if (mode == 0) {
          outf[(size_t)m * N + n] = val;
        } else {
          const int bb = m >> 10, l = m & (Lc - 1);
          const int hh = n >> 6,  d = n & (DKc - 1);
          const size_t idx = (mode == 1)
              ? ((((size_t)bb * Hc + hh) * Lc + l) * DKc + d)
              : ((((size_t)bb * Hc + hh) * DKc + d) * Lc + l);
          outb[idx] = f2bf(val);
        }
      }
    }
  }
}

// ---------------------------------------------------------------------------
// One workgroup (4 waves) per (b,h, 16-row q tile).
// grid (L/16, B*H), block 128.
__global__ void __launch_bounds__(128)
attn_kernel(const u16* __restrict__ qg, const u16* __restrict__ kg,
            const u16* __restrict__ vtg, const u16* __restrict__ posg,
            u16* __restrict__ attnb) {
  __shared__ float S[16 * Lc];          // 64 KB score rows
  __shared__ float pacc[4][16 * DKc];   // 16 KB per-wave AV partials
  __shared__ float red[16 * 8];
  __shared__ float rowst[16];

  const int i0   = blockIdx.x * 16;
  const int bh   = blockIdx.y;
  const int bidx = bh / Hc, hidx = bh % Hc;
  const int tid  = threadIdx.x;
  const int wave = tid >> 5;
  const int lane = tid & 31;
  const int r    = lane & 15;
  const int hi   = lane >> 4;

  const u16* q  = qg  + (size_t)bh * Lc * DKc;   // [L, dk]
  const u16* kh = kg  + (size_t)bh * Lc * DKc;   // [L, dk]
  const u16* vt = vtg + (size_t)bh * DKc * Lc;   // [dk, L]

  // Q fragments for the 16-row tile (dk = 64 -> two K=32 steps), per wave.
  FragB aq[2];
#pragma unroll
  for (int s = 0; s < 2; s++)
    load_a(aq[s], q + (size_t)(i0 + r) * DKc, s * 32, hi);

  // ---- content scores: S[ii,j] = (q_i . k_j) / 8  (double-buffered) ----
  {
    FragB kcur[2], knxt[2];
    auto load_kt = [&](FragB* dst, int j0) {
#pragma unroll
      for (int s = 0; s < 2; s++)
        load_b16(dst[s], kh + (size_t)(j0 + r) * DKc, s * 32, hi);
    };
    load_kt(kcur, wave * 16);
    for (int jt = wave; jt < Lc / 16; jt += 8) {   // 16 tiles/wave, unroll 2
      load_kt(knxt, (jt + 4) * 16);
      v8f acc = {};
      acc = wmma_bf16(aq[0], kcur[0], acc);
      acc = wmma_bf16(aq[1], kcur[1], acc);
#pragma unroll
      for (int v = 0; v < 8; v++)
        S[(hi * 8 + v) * Lc + jt * 16 + r] = acc[v] * 0.125f;
      if (jt + 8 < Lc / 16) load_kt(kcur, (jt + 8) * 16);
      v8f acc2 = {};
      acc2 = wmma_bf16(aq[0], knxt[0], acc2);
      acc2 = wmma_bf16(aq[1], knxt[1], acc2);
#pragma unroll
      for (int v = 0; v < 8; v++)
        S[(hi * 8 + v) * Lc + (jt + 4) * 16 + r] = acc2[v] * 0.125f;
    }
  }
  __syncthreads();

  // ---- relative scores with skew:  t = j + 15 - ii,  pos row = base + t ----
  const int base = Lc - 16 - i0;   // >= 0
  for (int tt = wave; tt < Lc / 16 + 1; tt += 4) {   // 65 t-tiles (t in [0,1039])
    const int t0 = tt * 16;
    v8f acc = {};
#pragma unroll
    for (int s = 0; s < 2; s++) {
      FragB b;
      load_b16(b, posg + (size_t)(base + t0 + r) * DKc, s * 32, hi);
      acc = wmma_bf16(aq[s], b, acc);
    }
#pragma unroll
    for (int v = 0; v < 8; v++) {
      const int ii = hi * 8 + v;
      const int j  = t0 + r + ii - 15;       // unique (ii,j) per (ii,t): no races
      if (j >= 0 && j < Lc) S[ii * Lc + j] += acc[v];
    }
  }
  __syncthreads();

  // ---- softmax over each of the 16 rows (128 threads = 16 rows x 8 segments) ----
  {
    const int row = tid & 15, seg = tid >> 4;
    float* Sr = S + row * Lc;
    float mx = -3.0e38f;
    for (int c = seg * 128; c < seg * 128 + 128; c++) mx = fmaxf(mx, Sr[c]);
    red[row * 8 + seg] = mx;
    __syncthreads();
    if (tid < 16) {
      float m2 = red[tid * 8];
      for (int s2 = 1; s2 < 8; s2++) m2 = fmaxf(m2, red[tid * 8 + s2]);
      rowst[tid] = m2;
    }
    __syncthreads();
    const float rm = rowst[row];
    float sum = 0.f;
    for (int c = seg * 128; c < seg * 128 + 128; c++) {
      const float e = __expf(Sr[c] - rm);
      Sr[c] = e;
      sum += e;
    }
    red[row * 8 + seg] = sum;
    __syncthreads();
    if (tid < 16) {
      float t = 0.f;
      for (int s2 = 0; s2 < 8; s2++) t += red[tid * 8 + s2];
      rowst[tid] = 1.0f / t;
    }
    __syncthreads();
    const float inv = rowst[row];
    for (int c = seg * 128; c < seg * 128 + 128; c++) Sr[c] *= inv;
  }
  __syncthreads();

  // ---- out = A @ V : k split over 4 waves (256 each), double-buffered V ----
  {
    const u16* vtr0 = vt + (size_t)(0 * 16 + r) * Lc;
    const u16* vtr1 = vt + (size_t)(1 * 16 + r) * Lc;
    const u16* vtr2 = vt + (size_t)(2 * 16 + r) * Lc;
    const u16* vtr3 = vt + (size_t)(3 * 16 + r) * Lc;

    FragB va[4], vb[4], af, af2;
    auto load_vt = [&](FragB* dst, int ks) {
      const int kk = wave * 256 + ks * 32;
      load_b16(dst[0], vtr0, kk, hi);
      load_b16(dst[1], vtr1, kk, hi);
      load_b16(dst[2], vtr2, kk, hi);
      load_b16(dst[3], vtr3, kk, hi);
    };
    auto build_a = [&](FragB& f, int ks) {   // convert probabilities to bf16 frag
      const float* Sr = S + r * Lc + wave * 256 + ks * 32 + hi * 8;
#pragma unroll
      for (int e = 0; e < 8; e++) {
        f.s[e]     = f2bf(Sr[e]);
        f.s[e + 8] = f2bf(Sr[e + 16]);
      }
    };

    v8f acc4[4] = {};
    load_vt(va, 0);
    for (int ks = 0; ks < 8; ks += 2) {
      load_vt(vb, ks + 1);
      build_a(af, ks);
      acc4[0] = wmma_bf16(af, va[0], acc4[0]);
      acc4[1] = wmma_bf16(af, va[1], acc4[1]);
      acc4[2] = wmma_bf16(af, va[2], acc4[2]);
      acc4[3] = wmma_bf16(af, va[3], acc4[3]);
      if (ks + 2 < 8) load_vt(va, ks + 2);
      build_a(af2, ks + 1);
      acc4[0] = wmma_bf16(af2, vb[0], acc4[0]);
      acc4[1] = wmma_bf16(af2, vb[1], acc4[1]);
      acc4[2] = wmma_bf16(af2, vb[2], acc4[2]);
      acc4[3] = wmma_bf16(af2, vb[3], acc4[3]);
    }
#pragma unroll
    for (int nt = 0; nt < 4; nt++)
#pragma unroll
      for (int v = 0; v < 8; v++)
        pacc[wave][(hi * 8 + v) * DKc + nt * 16 + r] = acc4[nt][v];
  }
  __syncthreads();

  for (int e = tid; e < 16 * DKc; e += 128) {
    const float s4 = pacc[0][e] + pacc[1][e] + pacc[2][e] + pacc[3][e];
    const int ii = e >> 6, d = e & 63;
    attnb[((size_t)bidx * Lc + (i0 + ii)) * Dc + hidx * DKc + d] = f2bf(s4);
  }
}

// ---------------------------------------------------------------------------
extern "C" void kernel_launch(void* const* d_in, const int* in_sizes, int n_in,
                              void* d_out, int out_size, void* d_ws, size_t ws_size,
                              hipStream_t stream) {
  (void)in_sizes; (void)n_in; (void)out_size; (void)ws_size;

  const float* x   = (const float*)d_in[0];
  const float* Wq  = (const float*)d_in[1];
  const float* bq  = (const float*)d_in[2];
  const float* Wk  = (const float*)d_in[3];
  const float* bk  = (const float*)d_in[4];
  const float* Wv  = (const float*)d_in[5];
  const float* bv  = (const float*)d_in[6];
  const float* Wo  = (const float*)d_in[7];
  const float* bo  = (const float*)d_in[8];
  const float* pos = (const float*)d_in[9];
  float* out = (float*)d_out;

  const size_t NX  = (size_t)Bc * Lc * Dc;          // 4 Mi elems
  const size_t NW  = (size_t)Dc * Dc;               // 1 Mi
  const size_t NP  = (size_t)2048 * DKc;            // padded pos rows
  const size_t NPs = (size_t)(2 * Lc - 1) * DKc;    // real pos rows

  u16* p    = (u16*)d_ws;
  u16* xb   = p; p += NX;
  u16* wqb  = p; p += NW;
  u16* wkb  = p; p += NW;
  u16* wvb  = p; p += NW;
  u16* wob  = p; p += NW;
  u16* posb = p; p += NP;
  u16* qb2  = p; p += NX;
  u16* kb2  = p; p += NX;
  u16* vtb  = p; p += NX;
  u16* atb  = p; p += NX;                           // ~48 MiB total

  auto cast = [&](const float* s, u16* d, size_t ns, size_t nd) {
    cast_f32_to_bf16<<<dim3((unsigned)((nd + 255) / 256)), dim3(256), 0, stream>>>(
        s, d, (int)ns, (int)nd);
  };
  cast(x,   xb,   NX,  NX);
  cast(Wq,  wqb,  NW,  NW);
  cast(Wk,  wkb,  NW,  NW);
  cast(Wv,  wvb,  NW,  NW);
  cast(Wo,  wob,  NW,  NW);
  cast(pos, posb, NPs, NP);   // zero-pad row 2047

  const dim3 gg(Bc * Lc / 32, Dc / 64);   // (128, 16)
  gemm_bf16<<<gg, dim3(32), 0, stream>>>(xb, wqb, bq, nullptr, qb2, Bc * Lc, Dc, Dc, 1);
  gemm_bf16<<<gg, dim3(32), 0, stream>>>(xb, wkb, bk, nullptr, kb2, Bc * Lc, Dc, Dc, 1);
  gemm_bf16<<<gg, dim3(32), 0, stream>>>(xb, wvb, bv, nullptr, vtb, Bc * Lc, Dc, Dc, 2);

  attn_kernel<<<dim3(Lc / 16, Bc * Hc), dim3(128), 0, stream>>>(qb2, kb2, vtb, posb, atb);

  gemm_bf16<<<gg, dim3(32), 0, stream>>>(atb, wob, bo, out, nullptr, Bc * Lc, Dc, Dc, 0);
}
